// TkStateSpaceModule_1322849927768
// MI455X (gfx1250) — matrix operand order repaired and checked by reading, not compiled, not run on hardware
//
#include <hip/hip_runtime.h>
#include <math.h>

typedef __attribute__((ext_vector_type(16))) _Float16 v16h;
typedef __attribute__((ext_vector_type(8)))  _Float16 v8h;
typedef __attribute__((ext_vector_type(4)))  _Float16 v4h;
typedef __attribute__((ext_vector_type(8)))  float    v8f;
typedef _Float16 half_t;

#define D_IN   256
#define D_ST   256
#define D_OUT  256
#define BATCH  8
#define SEQ    4096
#define BS     (BATCH * SEQ)      // 32768 rows
#define CHUNK  128
#define NCHUNK (SEQ / CHUNK)      // 32

// ---------------------------------------------------------------------------
// Async global->LDS b128 copy (CDNA5 GLOBAL_LOAD_ASYNC_TO_LDS_B128),
// tracked by ASYNCcnt; drained with s_wait_asynccnt <= imm.
// ---------------------------------------------------------------------------
__device__ __forceinline__ void async_ldg_b128(unsigned lds_byte_addr, const void* gptr)
{
    asm volatile("global_load_async_to_lds_b128 %0, %1, off"
                 :
                 : "v"(lds_byte_addr), "v"((unsigned long long)(size_t)gptr)
                 : "memory");
}

template <int CNT>
__device__ __forceinline__ void wait_async_le()
{
    asm volatile("s_wait_asynccnt %0" :: "n"(CNT) : "memory");
}

// ---------------------------------------------------------------------------
// Convert x (f32) -> xh (f16) once: halves A-tile HBM traffic and lets the
// GEMM stage A tiles with async b128 copies (no conversion in the hot loop).
// ---------------------------------------------------------------------------
__global__ __launch_bounds__(256) void cvt_to_f16(
    const float* __restrict__ x, half_t* __restrict__ xh)
{
    int i = (blockIdx.x * 256 + threadIdx.x) * 4;
    float4 v = *(const float4*)(x + i);
    v4h o;
    o.x = (half_t)v.x; o.y = (half_t)v.y; o.z = (half_t)v.z; o.w = (half_t)v.w;
    *(v4h*)(xh + i) = o;
}

// ---------------------------------------------------------------------------
// Setup 1: discretize poles; emit TRANSPOSED packed input weight
//   W0t[n][k]      = real(B_bar[n][k])   n in [0,256)
//   W0t[256+n][k]  = imag(B_bar[n][k])
// plus A_bar and A_bar^CHUNK.
// ---------------------------------------------------------------------------
__global__ __launch_bounds__(256) void ssm_setup1(
    const float* __restrict__ log_lambda_real,
    const float* __restrict__ lambda_imag,
    const float* __restrict__ log_dt,
    const float* __restrict__ Bmat,
    float* __restrict__ Abar_r, float* __restrict__ Abar_i,
    float* __restrict__ A128_r, float* __restrict__ A128_i,
    half_t* __restrict__ W0t)
{
    int n = threadIdx.x;
    float dt = __expf(log_dt[n]);
    dt = fminf(fmaxf(dt, 0.005f), 0.1f);
    float Lr = -__expf(log_lambda_real[n]);
    float Li = lambda_imag[n];
    float dr = 1.0f - 0.5f * dt * Lr;
    float di = -0.5f * dt * Li;
    float inv = 1.0f / (dr * dr + di * di);
    float nr = 1.0f + 0.5f * dt * Lr;
    float ni = 0.5f * dt * Li;
    float Ar = (nr * dr + ni * di) * inv;
    float Ai = (ni * dr - nr * di) * inv;
    Abar_r[n] = Ar;
    Abar_i[n] = Ai;
    float pr = Ar, pi = Ai;
#pragma unroll
    for (int t = 0; t < 7; ++t) {           // A_bar^128
        float qr = pr * pr - pi * pi;
        float qi = 2.0f * pr * pi;
        pr = qr; pi = qi;
    }
    A128_r[n] = pr;
    A128_i[n] = pi;
    float fr =  dt * dr * inv;              // dt / denom
    float fi = -dt * di * inv;
    for (int k = 0; k < D_IN; ++k) {
        float b = Bmat[n * D_IN + k];
        W0t[(size_t)n * D_IN + k]         = (half_t)(fr * b);
        W0t[(size_t)(n + 256) * D_IN + k] = (half_t)(fi * b);
    }
}

// ---------------------------------------------------------------------------
// Setup 2: TRANSPOSED packed output weight  W2t[o][k]:
//   k<256 -> C_r[o][k] ;  k>=256 -> -C_i[o][k-256]
// so y = [h_r | h_i] @ W2 = h_r C_r^T - h_i C_i^T.
// ---------------------------------------------------------------------------
__global__ __launch_bounds__(256) void ssm_setup2(
    const float* __restrict__ C_real,
    const float* __restrict__ C_imag,
    half_t* __restrict__ W2t)
{
    int idx = blockIdx.x * 256 + threadIdx.x;   // 256*512 elements
    int o = idx >> 9;
    int k = idx & 511;
    float v = (k < 256) ? C_real[(size_t)o * D_ST + k]
                        : -C_imag[(size_t)o * D_ST + (k - 256)];
    W2t[(size_t)o * 512 + k] = (half_t)v;
}

// ---------------------------------------------------------------------------
// WMMA GEMM:  out[M x N](f32) = A[M x K](f16) * Wt[N x K](f16, transposed)
// K, N compile-time so store/row strides fold into instruction offsets.
// Block: 256 threads = 8 wave32s, tile 128(M) x 64(N), K step 32.
// Each wave computes a 2x2 grid of 16x16 WMMA tiles (4 v_wmma / K step).
// Double-buffered LDS: stage k+1 issued via GLOBAL_LOAD_ASYNC_TO_LDS_B128
// before waiting asynccnt<=3 for stage k (async loads complete in order),
// overlapping the tile fetch with a full WMMA stage.
// Fragments read with ds_load_b128 (ISA 7.12.2 layouts, stride 40 halves).
// ---------------------------------------------------------------------------
template <int K, int N>
__global__ __launch_bounds__(256) void wmma_gemm(
    const half_t* __restrict__ A,
    const half_t* __restrict__ Wt,
    float* __restrict__ out,
    int M)
{
    __shared__ half_t As[2][128 * 40];   // 128 rows x 32 halves, stride 40
    __shared__ half_t Bs[2][64 * 40];    // 64 rows  x 32 halves, stride 40

    const int tid  = threadIdx.x;
    const int wave = tid >> 5;
    const int lane = tid & 31;
    const int hi   = lane >> 4;       // lane half
    const int mr   = lane & 15;       // A row in tile
    const int nc   = lane & 15;       // B/C col in tile
    const int wm   = wave & 3;        // M quadrant (32 rows)
    const int wn   = wave >> 2;       // N half (32 cols)

    const int m0 = blockIdx.x * 128;
    const int n0 = blockIdx.y * 64;

    // staging assignments (3 async b128 per thread per stage)
    const int arow = tid >> 1;            // 0..127
    const int ah   = (tid & 1) * 16;      // halves 0..15 / 16..31
    const int brow = tid >> 2;            // 0..63
    const int bh   = (tid & 3) * 8;       // 8-half chunk

    auto stage = [&](int k0, int p) {
        const half_t* ga = A + (size_t)(m0 + arow) * K + k0 + ah;
        unsigned la = (unsigned)(size_t)&As[p][arow * 40 + ah];
        async_ldg_b128(la,      ga);
        async_ldg_b128(la + 16, ga + 8);
        const half_t* gw = Wt + (size_t)(n0 + brow) * K + k0 + bh;
        unsigned lw = (unsigned)(size_t)&Bs[p][brow * 40 + bh];
        async_ldg_b128(lw, gw);
    };

    v8f acc[2][2] = {};

    constexpr int NK = K / 32;
    stage(0, 0);

    for (int kk = 0; kk < NK; ++kk) {
        const int p = kk & 1;
        if (kk + 1 < NK) {
            stage((kk + 1) * 32, p ^ 1);   // prefetch next tile (other buffer)
            wait_async_le<3>();            // stage kk's 3 ops have completed
        } else {
            wait_async_le<0>();
        }
        __syncthreads();                   // all waves' tiles for stage kk ready

        v16h a[2], b[2];
#pragma unroll
        for (int mt = 0; mt < 2; ++mt) {
            const half_t* base = &As[p][(wm * 32 + mt * 16 + mr) * 40];
            v8h lo  = *(const v8h*)(base + hi * 8);       // K = hi*8 .. +7
            v8h hi8 = *(const v8h*)(base + 16 + hi * 8);  // K = 16+hi*8 .. +7
            a[mt] = __builtin_shufflevector(lo, hi8,
                0, 1, 2, 3, 4, 5, 6, 7, 8, 9, 10, 11, 12, 13, 14, 15);
        }
#pragma unroll
        for (int nt = 0; nt < 2; ++nt) {
            const half_t* base = &Bs[p][(wn * 32 + nt * 16 + nc) * 40];
            v8h lo  = *(const v8h*)(base + hi * 8);       // K = hi*8 .. +7
            v8h hi8 = *(const v8h*)(base + 16 + hi * 8);  // K = 16+hi*8 .. +7
            b[nt] = __builtin_shufflevector(lo, hi8,
                0, 1, 2, 3, 4, 5, 6, 7, 8, 9, 10, 11, 12, 13, 14, 15);
        }

#pragma unroll
        for (int mt = 0; mt < 2; ++mt)
#pragma unroll
            for (int nt = 0; nt < 2; ++nt)
                acc[mt][nt] = __builtin_amdgcn_wmma_f32_16x16x32_f16(
                    false, a[mt], false, b[nt],
                    (short)0, acc[mt][nt], false, false);

        __syncthreads();   // all waves done reading buffer p before reuse
    }

    // ---- store: one base address; constexpr N folds offsets into immediates
    float* ob = out + (size_t)(m0 + wm * 32 + 8 * hi) * N + n0 + wn * 32 + nc;
#pragma unroll
    for (int mt = 0; mt < 2; ++mt)
#pragma unroll
        for (int nt = 0; nt < 2; ++nt)
#pragma unroll
            for (int r = 0; r < 8; ++r)
                ob[(size_t)(mt * 16 + r) * N + nt * 16] = acc[mt][nt][r];
}

// ---------------------------------------------------------------------------
// Scan pass A: chunk-final local state from zero init (one thread per n).
// ---------------------------------------------------------------------------
__global__ __launch_bounds__(256) void ssm_chunk_final(
    const float* __restrict__ u,
    const float* __restrict__ Abar_r, const float* __restrict__ Abar_i,
    float* __restrict__ cfr, float* __restrict__ cfi)
{
    int b = blockIdx.x / NCHUNK;
    int c = blockIdx.x % NCHUNK;
    int n = threadIdx.x;
    float Ar = Abar_r[n], Ai = Abar_i[n];
    float hr = 0.0f, hh = 0.0f;
    const float* ub = u + (size_t)(b * SEQ + c * CHUNK) * 512;
    for (int j = 0; j < CHUNK; ++j) {
        float ur = ub[(size_t)j * 512 + n];
        float ui = ub[(size_t)j * 512 + 256 + n];
        float nr = Ar * hr - Ai * hh + ur;
        float ni = Ar * hh + Ai * hr + ui;
        hr = nr; hh = ni;
    }
    cfr[(size_t)(b * NCHUNK + c) * 256 + n] = hr;
    cfi[(size_t)(b * NCHUNK + c) * 256 + n] = hh;
}

// ---------------------------------------------------------------------------
// Scan pass B: scan the 32 chunk carries, one thread per (b, n).
// ---------------------------------------------------------------------------
__global__ __launch_bounds__(256) void ssm_carry_scan(
    const float* __restrict__ cfr, const float* __restrict__ cfi,
    const float* __restrict__ A128_r, const float* __restrict__ A128_i,
    float* __restrict__ cir, float* __restrict__ cii)
{
    int b = blockIdx.x;
    int n = threadIdx.x;
    float Ar = A128_r[n], Ai = A128_i[n];
    float sr = 0.0f, si = 0.0f;
    for (int c = 0; c < NCHUNK; ++c) {
        size_t idx = (size_t)(b * NCHUNK + c) * 256 + n;
        cir[idx] = sr;
        cii[idx] = si;
        float fr = cfr[idx], fi = cfi[idx];
        float nr = Ar * sr - Ai * si + fr;
        float ni = Ar * si + Ai * sr + fi;
        sr = nr; si = ni;
    }
}

// ---------------------------------------------------------------------------
// Scan pass C: rerun local scan with carry-in; emit h as f16 [BS x 512].
// ---------------------------------------------------------------------------
__global__ __launch_bounds__(256) void ssm_final_scan(
    const float* __restrict__ u,
    const float* __restrict__ Abar_r, const float* __restrict__ Abar_i,
    const float* __restrict__ cir, const float* __restrict__ cii,
    half_t* __restrict__ hbuf)
{
    int b = blockIdx.x / NCHUNK;
    int c = blockIdx.x % NCHUNK;
    int n = threadIdx.x;
    float Ar = Abar_r[n], Ai = Abar_i[n];
    size_t cidx = (size_t)(b * NCHUNK + c) * 256 + n;
    float hr = cir[cidx], hh = cii[cidx];
    const float* ub = u + (size_t)(b * SEQ + c * CHUNK) * 512;
    half_t* hb = hbuf + (size_t)(b * SEQ + c * CHUNK) * 512;
    for (int j = 0; j < CHUNK; ++j) {
        float ur = ub[(size_t)j * 512 + n];
        float ui = ub[(size_t)j * 512 + 256 + n];
        float nr = Ar * hr - Ai * hh + ur;
        float ni = Ar * hh + Ai * hr + ui;
        hr = nr; hh = ni;
        hb[(size_t)j * 512 + n]       = (half_t)hr;
        hb[(size_t)j * 512 + 256 + n] = (half_t)hh;
    }
}

// ---------------------------------------------------------------------------
// Host-side orchestration
// ---------------------------------------------------------------------------
extern "C" void kernel_launch(void* const* d_in, const int* in_sizes, int n_in,
                              void* d_out, int out_size, void* d_ws, size_t ws_size,
                              hipStream_t stream)
{
    (void)in_sizes; (void)n_in; (void)out_size; (void)ws_size;

    const float* x    = (const float*)d_in[0];   // [8,4096,256]
    const float* llr  = (const float*)d_in[1];   // [256]
    const float* limg = (const float*)d_in[2];   // [256]
    const float* ldt  = (const float*)d_in[3];   // [256]
    const float* Bm   = (const float*)d_in[4];   // [256,256]
    const float* Cr   = (const float*)d_in[5];   // [256,256]
    const float* Ci   = (const float*)d_in[6];   // [256,256]
    float* y = (float*)d_out;                    // [8,4096,256] f32

    char* base = (char*)d_ws;
    size_t off = 0;
    auto carve = [&](size_t bytes) -> char* {
        char* p = base + off;
        off = (off + bytes + 255) & ~(size_t)255;
        return p;
    };
    float*  Abar_r = (float*)carve(256 * sizeof(float));
    float*  Abar_i = (float*)carve(256 * sizeof(float));
    float*  A128_r = (float*)carve(256 * sizeof(float));
    float*  A128_i = (float*)carve(256 * sizeof(float));
    half_t* W0t    = (half_t*)carve((size_t)512 * 256 * sizeof(half_t));
    half_t* W2t    = (half_t*)carve((size_t)256 * 512 * sizeof(half_t));
    float*  cfr    = (float*)carve((size_t)BATCH * NCHUNK * 256 * sizeof(float));
    float*  cfi    = (float*)carve((size_t)BATCH * NCHUNK * 256 * sizeof(float));
    float*  cir    = (float*)carve((size_t)BATCH * NCHUNK * 256 * sizeof(float));
    float*  cii    = (float*)carve((size_t)BATCH * NCHUNK * 256 * sizeof(float));
    half_t* xh     = (half_t*)carve((size_t)BS * D_IN * sizeof(half_t));
    half_t* hbuf   = (half_t*)carve((size_t)BS * 512 * sizeof(half_t));
    float*  u      = (float*)carve((size_t)BS * 512 * sizeof(float));

    // 0) x -> f16
    cvt_to_f16<<<(BS * D_IN) / (256 * 4), 256, 0, stream>>>(x, xh);

    // 1) discretization + packed transposed weights
    ssm_setup1<<<1, 256, 0, stream>>>(llr, limg, ldt, Bm,
                                      Abar_r, Abar_i, A128_r, A128_i, W0t);
    ssm_setup2<<<512, 256, 0, stream>>>(Cr, Ci, W2t);

    // 2) input projection: u[BS x 512] = xh[BS x 256] * W0t^T
    wmma_gemm<256, 512><<<dim3(BS / 128, 512 / 64), 256, 0, stream>>>(
        xh, W0t, u, BS);

    // 3) chunked complex recurrence scan
    ssm_chunk_final<<<BATCH * NCHUNK, 256, 0, stream>>>(u, Abar_r, Abar_i, cfr, cfi);
    ssm_carry_scan<<<BATCH, 256, 0, stream>>>(cfr, cfi, A128_r, A128_i, cir, cii);
    ssm_final_scan<<<BATCH * NCHUNK, 256, 0, stream>>>(u, Abar_r, Abar_i, cir, cii, hbuf);

    // 4) output projection: y[BS x 256] = hbuf[BS x 512] * W2t^T
    wmma_gemm<512, 256><<<dim3(BS / 128, 256 / 64), 256, 0, stream>>>(
        hbuf, W2t, y, BS);
}
